// SNNEncoder_979252543927
// MI455X (gfx1250) — compile-verified, matrix operand order checked by reading, not run on hardware
//
#include <hip/hip_runtime.h>

typedef __attribute__((ext_vector_type(16))) _Float16 v16h;
typedef __attribute__((ext_vector_type(8)))  _Float16 v8h;
typedef __attribute__((ext_vector_type(8)))  float    v8f;

#define LIF_ALPHA 0.92004441462932323f   // exp(-1/12)
#define LIF_VTH   0.9f

// ---------------------------------------------------------------------------
// fp32 -> fp16 conversion (grid-stride)
// ---------------------------------------------------------------------------
__global__ void cvt_f32_to_f16(const float* __restrict__ in,
                               _Float16* __restrict__ out, int n) {
  int i = blockIdx.x * blockDim.x + threadIdx.x;
  int stride = blockDim.x * gridDim.x;
  for (; i < n; i += stride) out[i] = (_Float16)in[i];
}

// ---------------------------------------------------------------------------
// WMMA fragment loader for a K-major (row-major MxK / NxK) f16 matrix.
// CDNA5 16-bit A 16x32 layout: lane l<16 holds row l, K {kk..kk+7, kk+16..kk+23};
// lane l>=16 holds row l-16, K {kk+8..kk+15, kk+24..kk+31}. B (32x16) from a
// row-major NxK matrix has the identical per-lane pattern.
// ---------------------------------------------------------------------------
__device__ __forceinline__ v16h load_frag_kmajor(const _Float16* __restrict__ base,
                                                 int row_base, int kk, int ldk,
                                                 int lane) {
  const int l16 = lane & 15;
  const int hi  = lane >> 4;                         // 0 or 1
  const _Float16* p = base + (size_t)(row_base + l16) * ldk + kk + hi * 8;
  union { v16h v; v8h h[2]; } u;
  u.h[0] = *(const v8h*)(p);        // K = kk + hi*8      .. +7   (global_load_b128)
  u.h[1] = *(const v8h*)(p + 16);   // K = kk + 16 + hi*8 .. +7
  return u.v;
}

// ---------------------------------------------------------------------------
// Fused GEMM (f16 WMMA, f32 acc) + LIF scan.
// One workgroup = one batch row b, one 64-wide slice of N, ALL 256 timesteps.
//   C[t, n] = bias[n] + sum_k A[b*256+t, k] * W[n, k]        (WMMA phase)
//   LIF scan over t in LDS                                    (scan phase)
// MODE 0: emit binary spikes z (f16) -> z_out[(b*256+t)*n_total + n]
// MODE 1: emit mean of post-reset mem over last 51 steps -> mean_out[b*n_total+n]
// ---------------------------------------------------------------------------
template <int K_DIM, int MODE>
__global__ __launch_bounds__(256)
void lif_layer_kernel(const _Float16* __restrict__ A,   // (B*256, K_DIM)
                      const _Float16* __restrict__ W,   // (n_total, K_DIM)
                      const float*    __restrict__ bias,
                      _Float16*       __restrict__ z_out,
                      float*          __restrict__ mean_out,
                      int n_total) {
  constexpr int TT = 256;   // timesteps == M tile
  constexpr int NT = 64;    // N tile
  __shared__ float cbuf[TT * NT];                     // 64 KB of the 320 KB LDS

  const int b     = blockIdx.x;
  const int nbase = blockIdx.y * NT;
  const int tid   = threadIdx.x;
  const int lane  = tid & 31;
  const int wave  = tid >> 5;        // 8 waves
  const int wm    = wave & 3;        // M rows [64*wm, 64*wm+64)
  const int wn    = wave >> 1 >> 1;  // N cols [32*wn, 32*wn+32)

  const _Float16* Ab = A + (size_t)b * TT * K_DIM;
  const _Float16* Wb = W + (size_t)nbase * K_DIM;

  const v8f vzero = {0.f, 0.f, 0.f, 0.f, 0.f, 0.f, 0.f, 0.f};
  v8f acc[4][2];
#pragma unroll
  for (int ms = 0; ms < 4; ++ms) {
#pragma unroll
    for (int ns = 0; ns < 2; ++ns) acc[ms][ns] = vzero;
  }

  for (int kk = 0; kk < K_DIM; kk += 32) {
    v16h af[4], bf[2];
#pragma unroll
    for (int ms = 0; ms < 4; ++ms)
      af[ms] = load_frag_kmajor(Ab, 64 * wm + 16 * ms, kk, K_DIM, lane);
#pragma unroll
    for (int ns = 0; ns < 2; ++ns)
      bf[ns] = load_frag_kmajor(Wb, 32 * wn + 16 * ns, kk, K_DIM, lane);

    if (kk + 64 < K_DIM) {  // pull next K-slab toward L0/L2 (global_prefetch_b8)
      __builtin_prefetch(Ab + (size_t)(64 * wm + (lane & 15)) * K_DIM + kk + 64, 0, 1);
      __builtin_prefetch(Wb + (size_t)(32 * wn + (lane & 15)) * K_DIM + kk + 64, 0, 1);
    }

#pragma unroll
    for (int ms = 0; ms < 4; ++ms) {
#pragma unroll
      for (int ns = 0; ns < 2; ++ns) {
        acc[ms][ns] = __builtin_amdgcn_wmma_f32_16x16x32_f16(
            false, af[ms], false, bf[ns], (short)0, acc[ms][ns], false, false);
      }
    }
  }

  // Epilogue: C (+bias) -> LDS.  (vgpr r, lane l) holds C[M = r + 8*(l>=16)][N = l%16]
  {
    const int l16 = lane & 15;
    const int hi  = lane >> 4;
#pragma unroll
    for (int ms = 0; ms < 4; ++ms) {
#pragma unroll
      for (int ns = 0; ns < 2; ++ns) {
        const int colL = 32 * wn + 16 * ns + l16;
        const float bv = bias[nbase + colL];
#pragma unroll
        for (int r = 0; r < 8; ++r) {
          const int row = 64 * wm + 16 * ms + r + 8 * hi;
          cbuf[row * NT + colL] = acc[ms][ns][r] + bv;
        }
      }
    }
  }
  __syncthreads();

  // LIF scan: 64 threads, each one column n; LDS reads are bank-conflict-free
  // (bank == n for every t).
  if (tid < NT) {
    const int n = tid;
    float mem = 0.f;
    if (MODE == 0) {
      _Float16* zp = z_out + (size_t)(b * TT) * n_total + nbase + n;
      for (int t = 0; t < TT; ++t) {
        const float a = cbuf[t * NT + n];
        mem = LIF_ALPHA * mem + a;
        const float z = (mem > LIF_VTH) ? 1.f : 0.f;
        mem -= LIF_VTH * z;
        zp[(size_t)t * n_total] = (_Float16)z;   // exact in f16
      }
    } else {
      float s = 0.f;
      for (int t = 0; t < TT; ++t) {
        const float a = cbuf[t * NT + n];
        mem = LIF_ALPHA * mem + a;
        const float z = (mem > LIF_VTH) ? 1.f : 0.f;
        mem -= LIF_VTH * z;                       // history stores post-reset mem
        if (t >= TT - 51) s += mem;
      }
      mean_out[(size_t)b * n_total + nbase + n] = s * (1.f / 51.f);
    }
  }
}

// ---------------------------------------------------------------------------
// Head: relu(lastk@Wr1^T+br1) @ Wr2^T + br2 -> LayerNorm -> L2 normalize.
// One workgroup per batch row; fp32 VALU, LDS reductions.
// ---------------------------------------------------------------------------
__global__ __launch_bounds__(256)
void head_kernel(const float* __restrict__ lastk,  // (64, 512)
                 const float* __restrict__ Wr1,    // (256, 512)
                 const float* __restrict__ br1,
                 const float* __restrict__ Wr2,    // (128, 256)
                 const float* __restrict__ br2,
                 const float* __restrict__ gamma,
                 const float* __restrict__ beta,
                 float* __restrict__ out) {        // (64, 128)
  __shared__ float sk[512];
  __shared__ float sh[256];
  __shared__ float se[128];
  __shared__ float rbuf[128];
  __shared__ float stat[2];
  const int b = blockIdx.x, tid = threadIdx.x;

  sk[tid]       = lastk[b * 512 + tid];
  sk[tid + 256] = lastk[b * 512 + tid + 256];
  __syncthreads();

  {  // h = relu(lastk @ Wr1^T + br1), 256 outputs, one per thread
    const float* w = Wr1 + (size_t)tid * 512;
    float acc = br1[tid];
    for (int i = 0; i < 512; ++i) acc = fmaf(sk[i], w[i], acc);
    sh[tid] = fmaxf(acc, 0.f);
  }
  __syncthreads();

  if (tid < 128) {  // emb = h @ Wr2^T + br2
    const float* w = Wr2 + (size_t)tid * 256;
    float acc = br2[tid];
    for (int i = 0; i < 256; ++i) acc = fmaf(sh[i], w[i], acc);
    se[tid] = acc;
  }
  __syncthreads();

  // mean
  if (tid < 128) rbuf[tid] = se[tid];
  __syncthreads();
  for (int s = 64; s > 0; s >>= 1) {
    if (tid < s) rbuf[tid] += rbuf[tid + s];
    __syncthreads();
  }
  if (tid == 0) stat[0] = rbuf[0] * (1.f / 128.f);
  __syncthreads();
  const float mu = stat[0];

  // variance (population)
  if (tid < 128) { const float d = se[tid] - mu; rbuf[tid] = d * d; }
  __syncthreads();
  for (int s = 64; s > 0; s >>= 1) {
    if (tid < s) rbuf[tid] += rbuf[tid + s];
    __syncthreads();
  }
  if (tid == 0) stat[1] = rbuf[0] * (1.f / 128.f);
  __syncthreads();
  const float rstd = rsqrtf(stat[1] + 1e-5f);

  if (tid < 128) {
    const float v = (se[tid] - mu) * rstd * gamma[tid] + beta[tid];
    se[tid] = v;
    rbuf[tid] = v * v;
  }
  __syncthreads();
  for (int s = 64; s > 0; s >>= 1) {
    if (tid < s) rbuf[tid] += rbuf[tid + s];
    __syncthreads();
  }
  if (tid == 0) stat[0] = fmaxf(sqrtf(rbuf[0]), 1e-12f);
  __syncthreads();

  if (tid < 128) out[b * 128 + tid] = se[tid] / stat[0];
}

// ---------------------------------------------------------------------------
extern "C" void kernel_launch(void* const* d_in, const int* in_sizes, int n_in,
                              void* d_out, int out_size, void* d_ws, size_t ws_size,
                              hipStream_t stream) {
  const float* spikes = (const float*)d_in[0];
  const float* W1     = (const float*)d_in[1];
  const float* b1     = (const float*)d_in[2];
  const float* W2     = (const float*)d_in[3];
  const float* b2     = (const float*)d_in[4];
  const float* Wr1    = (const float*)d_in[5];
  const float* br1    = (const float*)d_in[6];
  const float* Wr2    = (const float*)d_in[7];
  const float* br2    = (const float*)d_in[8];
  const float* gamma  = (const float*)d_in[9];
  const float* beta   = (const float*)d_in[10];
  (void)in_sizes; (void)n_in; (void)out_size; (void)ws_size;

  constexpr int B = 64, T = 256, D = 2048, H1 = 1024, H2 = 512;

  // workspace carve (256B-aligned): ~106 MB total
  char* ws = (char*)d_ws;
  size_t off = 0;
  auto carve = [&](size_t bytes) {
    void* p = ws + off;
    off += (bytes + 255) & ~(size_t)255;
    return p;
  };
  _Float16* sp16  = (_Float16*)carve((size_t)B * T * D * 2);   // 67 MB
  _Float16* w1h   = (_Float16*)carve((size_t)H1 * D * 2);      //  4 MB
  _Float16* w2h   = (_Float16*)carve((size_t)H2 * H1 * 2);     //  1 MB
  _Float16* z1    = (_Float16*)carve((size_t)B * T * H1 * 2);  // 32 MB
  float*    lastk = (float*)   carve((size_t)B * H2 * 4);      // 128 KB

  cvt_f32_to_f16<<<dim3(4096), dim3(256), 0, stream>>>(spikes, sp16, B * T * D);
  cvt_f32_to_f16<<<dim3(2048), dim3(256), 0, stream>>>(W1, w1h, H1 * D);
  cvt_f32_to_f16<<<dim3(1024), dim3(256), 0, stream>>>(W2, w2h, H2 * H1);

  // layer 1: 64 x 16 workgroups, fused GEMM(16384x1024, K=2048) + scan -> z1
  lif_layer_kernel<2048, 0><<<dim3(B, H1 / 64), dim3(256), 0, stream>>>(
      sp16, w1h, b1, z1, nullptr, H1);

  // layer 2: 64 x 8 workgroups, fused GEMM(16384x512, K=1024) + scan -> last-51 mean
  lif_layer_kernel<1024, 1><<<dim3(B, H2 / 64), dim3(256), 0, stream>>>(
      z1, w2h, b2, nullptr, lastk, H2);

  head_kernel<<<dim3(B), dim3(256), 0, stream>>>(lastk, Wr1, br1, Wr2, br2,
                                                 gamma, beta, (float*)d_out);
}
